// Sequence_Time_GRU_91027536871507
// MI455X (gfx1250) — matrix operand order repaired and checked by reading, not compiled, not run on hardware
//
#include <hip/hip_runtime.h>
#include <hip/hip_bf16.h>

// ---------------------------------------------------------------------------
// Time-GRU persistent-scan kernel for MI455X (gfx1250, wave32, WMMA).
//
// The recurrence is latency-bound (16384 dependent steps, ~4 GFLOP total,
// ~9 MB of streamed traffic vs 23.3 TB/s HBM). One workgroup (8 wave32s)
// owns the entire scan; all 15 weight matrices are resident in LDS as f16
// (277 KB of the WGP's 320 KB), padded 93->96 and stored column-major so a
// lane's WMMA B operand is two contiguous ds_load_b128s.
//
// A(16x32) carries the cell's input vector in rows 0-7 and the hidden vector
// in rows 8-15 (per-lane pointer select). One K-chained
// v_wmma_f32_16x16x32_f16 sequence per column tile yields x.W on D row 0
// (lanes 0-15) and h.W on D row 8 (lanes 16-31) -> branch-free tile loop.
// Tile indexing is scalar (SALU); all six B-tile ds_loads issue before the
// WMMA chain (sched_barrier) so LDS latency is overlapped; gates use
// v_rcp_f32-based sigmoid/tanh; x/t/ft are prefetched one step ahead and the
// next-step x store shares the gates barrier interval (4 barriers/step).
// ---------------------------------------------------------------------------

#define HH   93
#define HP   96
#define TIN  8192
#define TOUT 8192

typedef __attribute__((ext_vector_type(16))) _Float16 v16h;
typedef __attribute__((ext_vector_type(8)))  _Float16 v8h;
typedef __attribute__((ext_vector_type(8)))  float    v8f;

// LDS layout ---------------------------------------------------------------
static constexpr int MAT_H     = HP * HP;        // 9216 halfs per padded matrix
static constexpr int NMAT      = 7;              // matrices per cell
static constexpr int WGT_HALFS = 15 * MAT_H;     // 14 cell matrices + Wout
static constexpr int WGT_BYTES = WGT_HALFS * 2;  // 276480 B

// float region (offsets in floats, relative to WGT_BYTES)
static constexpr int F_BR   = 0;                 // 2*96 (brx+brh folded)
static constexpr int F_BZ   = 192;               // 2*96
static constexpr int F_BN   = 384;               // 2*96
static constexpr int F_BT   = 576;               // 2*96
static constexpr int F_WTT  = 768;               // 2*96
static constexpr int F_BOUT = 960;               // 96
static constexpr int F_H1   = 1056;              // 96  (fp32 hidden, cell 1)
static constexpr int F_H2   = 1152;              // 96  (fp32 hidden, cell 2)
static constexpr int F_Y    = 1248;              // 7*96 gate pre-activations
static constexpr int F_RED  = 1920;              // 2*256 reduction scratch
static constexpr int F_SCAL = 2432;              // inv_tmax, inv_ftmax (+pad)
static constexpr int F_TOT  = 2440;

static constexpr int HOFF_BYTES = WGT_BYTES + F_TOT * 4;    // 286240 (16B aligned)
static constexpr int SMEM_BYTES = HOFF_BYTES + 288 * 2;     // + xf16/h1f16/h2f16

struct Params {
  const float* x;   const float* t;   const float* ft;
  const float* W[NMAT];   // stacked (2,93,93): Wrx,Wrh,Wzx,Wzh,Wnx,Wnh,Wtx
  const float* Wout;      // (93,93)
  const float* brx; const float* brh;
  const float* bzx; const float* bzh;
  const float* bnx; const float* btx;
  const float* Wtt; const float* bout;
  float* out;             // (8192, 93)
};

// Fast sigmoid / tanh: one v_exp_f32 + one v_rcp_f32 (no IEEE divide chain).
static __device__ __forceinline__ float sigm(float v) {
  return __builtin_amdgcn_rcpf(1.0f + __expf(-v));
}
static __device__ __forceinline__ float tanh_fast(float v) {
  return 2.0f * sigm(2.0f * v) - 1.0f;
}

static __device__ __forceinline__ v8f wmma_acc(v16h a, v16h b, v8f c) {
  // D = A(16x32 f16) * B(32x16 f16) + C(16x16 f32)
  return __builtin_amdgcn_wmma_f32_16x16x32_f16(
      /*neg_a=*/false, a, /*neg_b=*/false, b,
      /*c_mod=*/(short)0, c, /*reuse_a=*/false, /*reuse_b=*/false);
}

// A operand (16-bit 16x32 layout): lane holds K = [8*hi..8*hi+7] ++ [16+8*hi..23+8*hi]
static __device__ __forceinline__ v16h load_vecA(const _Float16* p, int kt, int hi) {
  const v8h a0 = *(const v8h*)(p + kt * 32 + hi * 8);
  const v8h a1 = *(const v8h*)(p + kt * 32 + 16 + hi * 8);
  v16h r;
#pragma unroll
  for (int j = 0; j < 8; ++j) { r[j] = a0[j]; r[j + 8] = a1[j]; }
  return r;
}

// B operand (32x16 f16): lane n holds K = [8*hi..8*hi+7] ++ [16+8*hi..23+8*hi]; column-major storage
static __device__ __forceinline__ v16h load_tileB(const _Float16* colp, int kt, int hi) {
  const v8h b0 = *(const v8h*)(colp + kt * 32 + hi * 8);
  const v8h b1 = *(const v8h*)(colp + kt * 32 + 16 + hi * 8);
  v16h r;
#pragma unroll
  for (int j = 0; j < 8; ++j) { r[j] = b0[j]; r[j + 8] = b1[j]; }
  return r;
}

// One cell's 7 matvecs into y[m*96 + n]:
//   m even (Wrx,Wzx,Wnx,Wtx): y = X . W_m   (D row 0  -> lanes 0-15)
//   m odd  (Wrh,Wzh,Wnh):     y = H . W_m   (D row 8  -> lanes 16-31)
static __device__ __forceinline__ void run_phase(const _Float16* __restrict__ wgt_c,
                                                 const _Float16* __restrict__ Xp,
                                                 const _Float16* __restrict__ Hp,
                                                 float* __restrict__ y,
                                                 int wids, int lane) {
  const int hi = lane >> 4, nlo = lane & 15;
  // Rows 0-7 carry X, rows 8-15 carry H (lane L feeds row L%16).
  const _Float16* vp = (nlo < 8) ? Xp : Hp;
  v16h a[3];
#pragma unroll
  for (int kt = 0; kt < 3; ++kt) a[kt] = load_vecA(vp, kt, hi);

  const int laneColOff = nlo * HP;                 // per-lane part of B column
  for (int tile = wids; tile < 42; tile += 8) {    // scalar loop (wids is SGPR)
    const int m = tile / 6, nt = tile - m * 6;     // scalar m, nt
    const _Float16* colp = wgt_c + (m * MAT_H + nt * 16 * HP) + laneColOff;
    const v16h b0 = load_tileB(colp, 0, hi);
    const v16h b1 = load_tileB(colp, 1, hi);
    const v16h b2 = load_tileB(colp, 2, hi);
    // Keep all six ds_loads ahead of the WMMA chain so the s_wait_dscnt
    // sequence is staged (4/2/0) instead of one LDS round-trip per WMMA.
    __builtin_amdgcn_sched_barrier(0);
    v8f acc = {0.f, 0.f, 0.f, 0.f, 0.f, 0.f, 0.f, 0.f};
    acc = wmma_acc(a[0], b0, acc);
    acc = wmma_acc(a[1], b1, acc);
    acc = wmma_acc(a[2], b2, acc);
    // x-result lives on lanes 0-15 (row M=0), h-result on lanes 16-31 (row M=8)
    if (hi == (m & 1)) y[(m * HP + nt * 16) + nlo] = acc[0];
  }
}

static __device__ __forceinline__ void gates(const float* __restrict__ y,
                                             float* __restrict__ hf,
                                             _Float16* __restrict__ hf16,
                                             const float* br, const float* bz,
                                             const float* bn, const float* bt,
                                             const float* wtt,
                                             float dt, int i) {
  const float r  = sigm(y[i] + y[HP + i] + br[i]);
  const float z  = sigm(y[2 * HP + i] + y[3 * HP + i] + bz[i]);
  const float Tt = sigm(y[6 * HP + i] + bt[i] + sigm(dt * wtt[i]));
  const float n  = tanh_fast(y[4 * HP + i] + bn[i] + r * y[5 * HP + i]) * Tt;
  const float h  = hf[i];
  const float hn = (1.0f - z) * n + z * h;
  hf[i]   = hn;
  hf16[i] = (_Float16)hn;
}

__global__ __launch_bounds__(256) void timegru_scan(Params p) {
  extern __shared__ char smem[];
  _Float16* wgt = (_Float16*)smem;
  float*    fb  = (float*)(smem + WGT_BYTES);
  float* br  = fb + F_BR;   float* bz  = fb + F_BZ;  float* bn  = fb + F_BN;
  float* bt  = fb + F_BT;   float* wtt = fb + F_WTT; float* bo  = fb + F_BOUT;
  float* h1  = fb + F_H1;   float* h2  = fb + F_H2;  float* y   = fb + F_Y;
  float* red = fb + F_RED;  float* scal = fb + F_SCAL;
  _Float16* xf  = (_Float16*)(smem + HOFF_BYTES);
  _Float16* h1f = xf + HP;
  _Float16* h2f = xf + 2 * HP;

  const int tid  = threadIdx.x, wid = tid >> 5, lane = tid & 31;
  const int wids = __builtin_amdgcn_readfirstlane(wid);   // scalar wave id

  // ---- init: weights f32 -> f16, transposed (column-major) + zero-padded ----
  for (int idx = tid; idx < WGT_HALFS; idx += 256) {
    const int mi = idx / MAT_H;
    const int rr = idx - mi * MAT_H;
    const int n  = rr / HP;
    const int k  = rr - n * HP;
    float v = 0.f;
    if (k < HH && n < HH) {
      if (mi < 14) {
        const int c = mi / NMAT, m = mi - c * NMAT;
        v = p.W[m][c * HH * HH + k * HH + n];
      } else {
        v = p.Wout[k * HH + n];
      }
    }
    wgt[idx] = (_Float16)v;
  }
  // biases (fold brx+brh, bzx+bzh) and Wtt row, per cell
  for (int i = tid; i < 192; i += 256) {
    const int c = i / HP, j = i - c * HP;
    const bool ok = (j < HH);
    const int o = c * HH + j;
    br[i]  = ok ? (p.brx[o] + p.brh[o]) : 0.f;
    bz[i]  = ok ? (p.bzx[o] + p.bzh[o]) : 0.f;
    bn[i]  = ok ? p.bnx[o] : 0.f;
    bt[i]  = ok ? p.btx[o] : 0.f;
    wtt[i] = ok ? p.Wtt[o] : 0.f;
  }
  if (tid < HP) {
    bo[tid]  = (tid < HH) ? p.bout[tid] : 0.f;
    h1[tid]  = 0.f;  h2[tid] = 0.f;
    // Seed xf with x[0]; h vectors start at zero.
    xf[tid]  = (tid < HH) ? (_Float16)p.x[tid] : (_Float16)0.f;
    h1f[tid] = (_Float16)0.f;
    h2f[tid] = (_Float16)0.f;
  }
  // ---- t.max() / ft.max() reductions ----
  float mt = -1e30f, mf = -1e30f;
  for (int i = tid; i < TIN;  i += 256) mt = fmaxf(mt, p.t[i]);
  for (int i = tid; i < TOUT; i += 256) mf = fmaxf(mf, p.ft[i]);
  red[tid] = mt; red[256 + tid] = mf;
  __syncthreads();
  for (int s = 128; s > 0; s >>= 1) {
    if (tid < s) {
      red[tid]       = fmaxf(red[tid],       red[tid + s]);
      red[256 + tid] = fmaxf(red[256 + tid], red[256 + tid + s]);
    }
    __syncthreads();
  }
  if (tid == 0) { scal[0] = 1.0f / red[0]; scal[1] = 1.0f / red[256]; }
  __syncthreads();

  const _Float16* wgt1 = wgt;                 // cell-1 matrices
  const _Float16* wgt2 = wgt + 7 * MAT_H;     // cell-2 matrices
  const _Float16* wout = wgt + 14 * MAT_H;    // output projection

  const float invT = scal[0], invF = scal[1];

  // ================= input phase =================
  // xf already holds x[0]; prefetch x[s+1]/t[s+1] at step top so HBM latency
  // overlaps both WMMA phases; the xf store for s+1 shares the gates-2
  // barrier interval (disjoint LDS) -> 4 barriers per step.
  float tr = p.t[0];
  for (int s = 0; s < TIN; ++s) {
    const float dt = tr * invT;
    float xnext = 0.f;
    if (s + 1 < TIN) {                 // issue next-step loads early
      if (tid < HH) xnext = p.x[(s + 1) * HH + tid];
      tr = p.t[s + 1];
    }
    run_phase(wgt1, xf, h1f, y, wids, lane);
    __syncthreads();
    if (tid < HH) gates(y, h1, h1f, br, bz, bn, bt, wtt, dt, tid);
    __syncthreads();
    run_phase(wgt2, h1f, h2f, y, wids, lane);
    __syncthreads();
    if (tid < HH) gates(y, h2, h2f, br + HP, bz + HP, bn + HP, bt + HP, wtt + HP, dt, tid);
    if (tid < HP) xf[tid] = (tid < HH) ? (_Float16)xnext : (_Float16)0.f;
    __syncthreads();
  }

  // ================= output phase =================
  float fr = p.ft[0];
  for (int s = 0; s < TOUT; ++s) {
    const float dt = fr * invF;
    if (s + 1 < TOUT) fr = p.ft[s + 1];          // prefetch next scalar
    run_phase(wgt1, h2f, h1f, y, wids, lane);    // cell-1 input is h2
    __syncthreads();
    if (tid < HH) gates(y, h1, h1f, br, bz, bn, bt, wtt, dt, tid);
    __syncthreads();
    run_phase(wgt2, h1f, h2f, y, wids, lane);
    __syncthreads();
    if (tid < HH) gates(y, h2, h2f, br + HP, bz + HP, bn + HP, bt + HP, wtt + HP, dt, tid);
    __syncthreads();
    // out[s] = h2 @ Wout + bout  (6 N-tiles over waves 0..5; scalar branch so
    // WMMA never sits inside an exec-masked region)
    if (wids < 6) {
      const int hi = lane >> 4, nlo = lane & 15;
      const v16h a0 = load_vecA(h2f, 0, hi);
      const v16h a1 = load_vecA(h2f, 1, hi);
      const v16h a2 = load_vecA(h2f, 2, hi);
      const _Float16* colp = wout + (wids * 16 * HP) + nlo * HP;
      const v16h b0 = load_tileB(colp, 0, hi);
      const v16h b1 = load_tileB(colp, 1, hi);
      const v16h b2 = load_tileB(colp, 2, hi);
      __builtin_amdgcn_sched_barrier(0);
      v8f acc = {0.f, 0.f, 0.f, 0.f, 0.f, 0.f, 0.f, 0.f};
      acc = wmma_acc(a0, b0, acc);
      acc = wmma_acc(a1, b1, acc);
      acc = wmma_acc(a2, b2, acc);
      if (lane < 16) {
        const int col = wids * 16 + lane;
        if (col < HH) p.out[s * HH + col] = acc[0] + bo[col];
      }
    }
  }
}

extern "C" void kernel_launch(void* const* d_in, const int* in_sizes, int n_in,
                              void* d_out, int out_size, void* d_ws, size_t ws_size,
                              hipStream_t stream) {
  (void)in_sizes; (void)n_in; (void)out_size; (void)d_ws; (void)ws_size;
  Params p;
  p.x    = (const float*)d_in[0];
  p.t    = (const float*)d_in[1];
  p.ft   = (const float*)d_in[2];
  p.W[0] = (const float*)d_in[3];   // Wrx
  p.brx  = (const float*)d_in[4];
  p.W[1] = (const float*)d_in[5];   // Wrh
  p.brh  = (const float*)d_in[6];
  p.W[2] = (const float*)d_in[7];   // Wzx
  p.bzx  = (const float*)d_in[8];
  p.W[3] = (const float*)d_in[9];   // Wzh
  p.bzh  = (const float*)d_in[10];
  p.W[4] = (const float*)d_in[11];  // Wnx
  p.bnx  = (const float*)d_in[12];
  p.W[5] = (const float*)d_in[13];  // Wnh
  p.W[6] = (const float*)d_in[14];  // Wtx
  p.btx  = (const float*)d_in[15];
  p.Wtt  = (const float*)d_in[16];
  p.Wout = (const float*)d_in[17];
  p.bout = (const float*)d_in[18];
  p.out  = (float*)d_out;

  // Allow >64KB dynamic LDS (287 KB of the WGP's 320 KB).
  hipFuncSetAttribute((const void*)timegru_scan,
                      hipFuncAttributeMaxDynamicSharedMemorySize,
                      (int)SMEM_BYTES);
  timegru_scan<<<dim3(1), dim3(256), SMEM_BYTES, stream>>>(p);
}
